// MapProjection_10634339025376
// MI455X (gfx1250) — matrix operand order repaired and checked by reading, not compiled, not run on hardware
//
#include <hip/hip_runtime.h>
#include <hip/hip_fp16.h>

typedef __attribute__((ext_vector_type(16))) _Float16       v16h;
typedef __attribute__((ext_vector_type(8)))  _Float16       v8h;
typedef __attribute__((ext_vector_type(16))) unsigned short v16u;
typedef __attribute__((ext_vector_type(8)))  float          v8f;
typedef __attribute__((ext_vector_type(4)))  unsigned int   v4u;
typedef __attribute__((ext_vector_type(8)))  int            v8i;
typedef __attribute__((ext_vector_type(4)))  int            v4i;

#define N_   2
#define L_   512
#define D_   128
#define H_   64
#define K2   256
#define XB   8
#define YB   64
#define MTOT (N_ * L_ * L_)
#define WS_WEIGHT_OFF 1024

#ifndef USE_TDM
#if __has_builtin(__builtin_amdgcn_tensor_load_to_lds) && __has_builtin(__builtin_amdgcn_s_wait_tensorcnt)
#define USE_TDM 1
#else
#define USE_TDM 0
#endif
#endif

#if USE_TDM
__device__ __forceinline__ unsigned lds_addr_of(void* p) {
    return (unsigned)(unsigned long long)(__attribute__((address_space(3))) void*)p;
}
#endif

// ---------------------------------------------------------------------------
// Pass 0: zero per-channel stats; pre-swizzle conv_w (f32->f16) into the WMMA
// 16-bit A-matrix lane layout:  M = lane&15,  K = 16*(e>>3) + 8*(lane>>4) + (e&7)
// ---------------------------------------------------------------------------
__global__ void prep_kernel(const float* __restrict__ conv_w, float* __restrict__ ws) {
    int tid = threadIdx.x;
    if (tid < 2 * H_) ws[tid] = 0.0f;
    _Float16* wbuf = (_Float16*)((char*)ws + WS_WEIGHT_OFF);
    for (int i = tid; i < H_ * K2; i += 256) {
        int e     = i & 15;
        int lane  = (i >> 4) & 31;
        int ht    = (i >> 9) & 3;
        int chunk = i >> 11;
        int half  = lane >> 4;
        int m     = lane & 15;
        int h     = ht * 16 + m;
        int k     = 16 * (e >> 3) + 8 * half + (e & 7);
        wbuf[i] = (_Float16)conv_w[h * K2 + chunk * 32 + k];
    }
}

// ---------------------------------------------------------------------------
// Pass 1: fused pairwise features + WMMA GEMM + bias + ELU.
// Block = 256 thr = 8 waves, covers (n, 8 x, 64 y).  Wave (yt = w&3, xh = w>>2)
// computes a 16-wide y tile x all 64 h rows for 4 x values.
// z1 tile staged by TDM (f32, padded 132-dw rows), then converted once to a
// padded f16 tile; hot loop uses only packed-f16 VALU + v_wmma.
// ---------------------------------------------------------------------------
__global__ __launch_bounds__(256)
void conv_kernel(const float* __restrict__ z0, const float* __restrict__ z1,
                 const float* __restrict__ conv_b, float* __restrict__ out,
                 const float* __restrict__ ws) {
#if USE_TDM
    __shared__ float sZ1f[YB * 132];      // TDM destination (528B row stride)
#endif
    __shared__ _Float16 sZ1[YB * 136];    // f16 tile, 272B row stride
    __shared__ _Float16 sZ0[XB * D_];

    const int tid = threadIdx.x;
    const int n   = blockIdx.z;
    const int x0  = blockIdx.y * XB;
    const int y0  = blockIdx.x * YB;

#if USE_TDM
    if (tid < 32) {   // wave 0 issues the DMA for the z1 tile
        unsigned long long ga1 =
            (unsigned long long)(const void*)(z1 + ((size_t)(n * L_ + y0)) * D_);
        // group0: count=1 | lds_addr | global_addr[56:0] | type=2
        v4u g0a = { 1u, lds_addr_of(sZ1f), (unsigned)ga1,
                    (unsigned)((ga1 >> 32) & 0x01FFFFFFu) | 0x80000000u };
        // 1-D copy of 8192 f32: data_size=4B(2), pad_enable, 4 dw (code 3)
        // every 128 dw (code 6) -> LDS rows of 132 dwords
        int fl = (2 << 16) | (1 << 20) | (6 << 22) | (3 << 25);
        v8i g1a = { fl, (int)(8192u << 16), 0x00010000, (int)(8192u << 16),
                    0, 8192, 0, 0 };
        v4i zz = { 0, 0, 0, 0 };
#if __clang_major__ >= 23
        v8i zz8 = { 0, 0, 0, 0, 0, 0, 0, 0 };
        __builtin_amdgcn_tensor_load_to_lds(g0a, g1a, zz, zz, zz8, 0);
#else
        __builtin_amdgcn_tensor_load_to_lds(g0a, g1a, zz, zz, 0);
#endif
        __builtin_amdgcn_s_wait_tensorcnt(0);
    }
    // stage z0 rows (2 KB) directly as f16
    for (int i = tid; i < XB * D_; i += 256)
        sZ0[i] = (_Float16)z0[((size_t)(n * L_ + x0)) * D_ + i];
    __syncthreads();
    // one-shot convert: f32 padded tile -> f16 padded tile
    for (int i = tid; i < YB * D_; i += 256) {
        int yy = i >> 7, c = i & 127;
        sZ1[yy * 136 + c] = (_Float16)sZ1f[yy * 132 + c];
    }
#else
    for (int i = tid; i < YB * D_; i += 256) {
        int yy = i >> 7, c = i & 127;
        sZ1[yy * 136 + c] = (_Float16)z1[((size_t)(n * L_ + y0 + yy)) * D_ + c];
    }
    for (int i = tid; i < XB * D_; i += 256)
        sZ0[i] = (_Float16)z0[((size_t)(n * L_ + x0)) * D_ + i];
#endif
    __syncthreads();

    const int wave = tid >> 5, lane = tid & 31;
    const int half = lane >> 4, m = lane & 15;
    const int yt = wave & 3, xh = wave >> 2;
    const int ylocal = yt * 16 + m;
    const int yglob  = y0 + ylocal;

    // preload all A tiles once per wave (32 KB, L2-resident)
    const v16h* wb = (const v16h*)((const char*)ws + WS_WEIGHT_OFF);
    v16h areg[32];
#pragma unroll
    for (int t = 0; t < 32; ++t) areg[t] = wb[t * 32 + lane];

    float bias[4][8];
#pragma unroll
    for (int ht = 0; ht < 4; ++ht)
#pragma unroll
        for (int r = 0; r < 8; ++r)
            bias[ht][r] = conv_b[ht * 16 + half * 8 + r];

    for (int xi = 0; xi < 4; ++xi) {
        const int xx = xh * 4 + xi;
        v8f acc[4] = {};
#pragma unroll
        for (int chunk = 0; chunk < 8; ++chunk) {
            // B layout: element e of lane -> K = 16*(e>>3) + 8*half + (e&7), column = m
            const int cbase = (chunk & 3) * 32 + half * 8;    // c' in [0,128)
            const _Float16* pz1 = &sZ1[ylocal * 136 + cbase]; // 16B aligned
            const _Float16* pz0 = &sZ0[xx * D_ + cbase];      // 16B aligned
            union U16 { v16h v; v8h p[2]; v16u u; };
            U16 zb, za, bm;
            zb.p[0] = *(const v8h*)pz1;  zb.p[1] = *((const v8h*)pz1 + 2);
            za.p[0] = *(const v8h*)pz0;  za.p[1] = *((const v8h*)pz0 + 2);
            if (chunk < 4) {                 // |z0 - z1| channels (pk_sub + mask)
                bm.v = za.v - zb.v;
                bm.u = bm.u & (unsigned short)0x7FFF;
            } else {                         // z0 * z1 channels (pk_mul)
                bm.v = za.v * zb.v;
            }
#pragma unroll
            for (int ht = 0; ht < 4; ++ht)
                acc[ht] = __builtin_amdgcn_wmma_f32_16x16x32_f16(
                    false, areg[chunk * 4 + ht], false, bm.v, (short)0, acc[ht],
                    false, false);
        }
        // epilogue: bias + ELU + store.  D layout: VGPR r -> h = ht*16 + half*8 + r
        const int x = x0 + xx;
#pragma unroll
        for (int ht = 0; ht < 4; ++ht)
#pragma unroll
            for (int r = 0; r < 8; ++r) {
                int   h = ht * 16 + half * 8 + r;
                float v = acc[ht][r] + bias[ht][r];
                v = v > 0.0f ? v : (__expf(v) - 1.0f);
                out[(((size_t)(n * H_ + h)) * L_ + x) * L_ + yglob] = v;
            }
    }
}

// ---------------------------------------------------------------------------
// Pass 1b: per-channel sum / sum^2 from d_out (L2-resident, 192 MB L2 > 134 MB)
// Each block reduces a 65536-element slice of one channel.
// ---------------------------------------------------------------------------
__global__ __launch_bounds__(256)
void stats_kernel(const float* __restrict__ out, float* __restrict__ ws) {
    __shared__ float sS[8], sQ[8];
    size_t base = (size_t)blockIdx.x * 65536;
    int h = (int)((base >> 18) & 63);
    const float4* p = (const float4*)(out + base);
    float s = 0.0f, q = 0.0f;
    for (int i = threadIdx.x; i < 16384; i += 256) {
        float4 v = p[i];
        s += v.x + v.y + v.z + v.w;
        q += v.x * v.x + v.y * v.y + v.z * v.z + v.w * v.w;
    }
#pragma unroll
    for (int off = 1; off < 32; off <<= 1) {
        s += __shfl_xor(s, off, 32);
        q += __shfl_xor(q, off, 32);
    }
    int wave = threadIdx.x >> 5, lane = threadIdx.x & 31;
    if (lane == 0) { sS[wave] = s; sQ[wave] = q; }
    __syncthreads();
    if (threadIdx.x == 0) {
        float ts = 0.0f, tq = 0.0f;
#pragma unroll
        for (int w = 0; w < 8; ++w) { ts += sS[w]; tq += sQ[w]; }
        atomicAdd(&ws[h], ts);
        atomicAdd(&ws[H_ + h], tq);
    }
}

// ---------------------------------------------------------------------------
// Pass 2: batchnorm finalize, float4-vectorized, in place on d_out.
// ---------------------------------------------------------------------------
__global__ __launch_bounds__(256)
void norm_kernel(float* __restrict__ out, const float* __restrict__ ws,
                 const float* __restrict__ gamma, const float* __restrict__ beta) {
    size_t idx = (size_t)blockIdx.x * blockDim.x + threadIdx.x;
    size_t e0  = idx * 4;
    int h = (int)((e0 >> 18) & 63);
    const float inv = 1.0f / (float)MTOT;
    float mean = ws[h] * inv;
    float var  = ws[H_ + h] * inv - mean * mean;
    float rstd = rsqrtf(var + 1e-5f);
    float g = gamma[h] * rstd;
    float b = beta[h] - mean * g;
    float4 v = *(float4*)(out + e0);
    v.x = v.x * g + b;  v.y = v.y * g + b;
    v.z = v.z * g + b;  v.w = v.w * g + b;
    *(float4*)(out + e0) = v;
}

// ---------------------------------------------------------------------------
extern "C" void kernel_launch(void* const* d_in, const int* in_sizes, int n_in,
                              void* d_out, int out_size, void* d_ws, size_t ws_size,
                              hipStream_t stream) {
    (void)in_sizes; (void)n_in; (void)out_size; (void)ws_size;
    const float* z0     = (const float*)d_in[0];
    const float* z1     = (const float*)d_in[1];
    const float* conv_w = (const float*)d_in[2];
    const float* conv_b = (const float*)d_in[3];
    const float* gamma  = (const float*)d_in[4];
    const float* beta   = (const float*)d_in[5];
    float* out = (float*)d_out;
    float* ws  = (float*)d_ws;

    prep_kernel<<<1, 256, 0, stream>>>(conv_w, ws);
    conv_kernel<<<dim3(L_ / YB, L_ / XB, N_), 256, 0, stream>>>(z0, z1, conv_b, out, ws);
    stats_kernel<<<(N_ * H_ * L_ * L_) / 65536, 256, 0, stream>>>(out, ws);
    norm_kernel<<<(N_ * H_ * L_ * L_ / 4) / 256, 256, 0, stream>>>(out, ws, gamma, beta);
}
